// DSFS_83897891160149
// MI455X (gfx1250) — compile-verified, matrix operand order checked by reading, not run on hardware
//
#include <hip/hip_runtime.h>
#include <hip/hip_bf16.h>
#include <math.h>

// ---------------------------------------------------------------------------
// 2D cross-scan Mamba for MI455X (gfx1250, wave32, WMMA).
//
// Direction flips commute with the channel GEMMs, so the input projection
// (M=8192,K=256,N=1024) and output projection (M=8192,K=512,N=256) run ONCE.
// All GEMMs use v_wmma_f32_16x16x32_bf16 with 16x64 (dbl: 16x48) per-wave tiles; operands
// are pre-packed bf16 so every fragment is two global_load_b128s.
// The serial selective scan stages its shared dbl rows into LDS with
// GLOBAL_LOAD_ASYNC_TO_LDS_B32 (ASYNCcnt) and keeps states in VGPRs.
// ---------------------------------------------------------------------------

typedef __bf16 bf16_t;
typedef __attribute__((ext_vector_type(16))) __bf16 v16bf;
typedef __attribute__((ext_vector_type(8)))  __bf16 v8bf;
typedef __attribute__((ext_vector_type(8)))  float  v8f;

#define L_SEQ   4096
#define D_IN    512      // D_INNER
#define NDIR    4
#define NB      2
#define NSTATE  16
#define DTR     16       // DT_RANK
#define DBLW    48       // DTR + 2*D_STATE

// Workspace layout (bytes)
#define OFF_XM      ((size_t)0)                       // fp32 [2*4096, 512]   16 MiB
#define OFF_GATE    (OFF_XM    + (size_t)16777216)    // fp32 [2*4096, 512]   16 MiB
#define OFF_XS32    (OFF_GATE  + (size_t)16777216)    // fp32 [8*4096, 512]   64 MiB
#define OFF_XS16    (OFF_XS32  + (size_t)67108864)    // bf16 [8*4096, 512]   32 MiB
#define OFF_DBL     (OFF_XS16  + (size_t)33554432)    // fp32 [8*4096, 48]     6 MiB
#define OFF_YALL    (OFF_DBL   + (size_t)6291456)     // fp32 [8*4096, 512]   64 MiB
#define OFF_YSUM    (OFF_YALL  + (size_t)67108864)    // bf16 [2*4096, 512]    8 MiB
#define OFF_XT      (OFF_YSUM  + (size_t)8388608)     // bf16 [2*4096, 256]    4 MiB
#define OFF_WINT    (OFF_XT    + (size_t)4194304)     // bf16 [1024, 256]    512 KiB
#define OFF_WXT     (OFF_WINT  + (size_t)524288)      // bf16 [48, 512]       48 KiB
#define OFF_WOUTT   (OFF_WXT   + (size_t)49152)       // bf16 [256, 512]     256 KiB

__device__ __forceinline__ float sigmoidf_(float v) {
  return 1.0f / (1.0f + __expf(-v));
}

// ---- WMMA 16-bit fragment load (ISA 7.12.2) -------------------------------
// A-frag: lane holds row M=lane&15; regs 0-3 hold K=kb..kb+7, regs 4-7 hold
// K=kb+16..kb+23 (kb = 8*(lane>=16)) -> two contiguous 16B loads per lane.
// B-frag mirrored with lane = column N (operand stored transposed [n][k]).
__device__ __forceinline__ v16bf load_frag16(const bf16_t* __restrict__ base,
                                             int row0, int ldk, int k0, int lane) {
  const bf16_t* p = base + (size_t)(row0 + (lane & 15)) * ldk + k0 + ((lane >> 4) << 3);
  v8bf lo = *(const v8bf*)p;          // K = kb .. kb+7
  v8bf hi = *(const v8bf*)(p + 16);   // K = kb+16 .. kb+23
  return __builtin_shufflevector(lo, hi, 0, 1, 2, 3, 4, 5, 6, 7,
                                 8, 9, 10, 11, 12, 13, 14, 15);
}

// ---------------------------------------------------------------------------
// Prep kernels: pack operands as bf16 in fragment-friendly order.
// ---------------------------------------------------------------------------
__global__ void k_prep_xT(const float* __restrict__ x, bf16_t* __restrict__ xT) {
  // x: (b, c, l) fp32  ->  xT: [(b*4096+l), c] bf16     (2*256*4096 elements)
  const size_t idx = (size_t)blockIdx.x * blockDim.x + threadIdx.x;
  const int l = (int)(idx & 4095);
  const int c = (int)((idx >> 12) & 255);
  const int b = (int)(idx >> 20);
  xT[((size_t)((b << 12) + l)) * 256 + c] = (bf16_t)x[idx];
}

__global__ void k_prep_wT(const float* __restrict__ W, bf16_t* __restrict__ WT,
                          int K, int N) {
  // W: [K, N] fp32 -> WT: [N, K] bf16
  const int idx = blockIdx.x * blockDim.x + threadIdx.x;
  if (idx >= K * N) return;
  const int k = idx / N;
  const int n = idx - k * N;
  WT[(size_t)n * K + k] = (bf16_t)W[idx];
}

// ---------------------------------------------------------------------------
// Kernel 1: xz0 = z0 @ W_in.  M=8192, K=256, N=1024 -> xm | gate.
// Per wave: one 16x64 strip (4 accumulators, shared A fragment).
// ---------------------------------------------------------------------------
__global__ void k_gemm_in(const bf16_t* __restrict__ xT, const bf16_t* __restrict__ W_inT,
                          float* __restrict__ xm, float* __restrict__ gate) {
  const int lane = threadIdx.x & 31;
  const int wave = threadIdx.x >> 5;
  const int task = blockIdx.x * 8 + wave;     // 512 Mtiles * 16 strips = 8192
  const int mt = task >> 4;
  const int ns = task & 15;
  const int m0 = mt << 4, n0 = ns << 6;

  v8f acc[4] = {};
  for (int k0 = 0; k0 < 256; k0 += 32) {
    const v16bf a = load_frag16(xT, m0, 256, k0, lane);
#pragma unroll
    for (int q = 0; q < 4; ++q) {
      const v16bf b = load_frag16(W_inT, n0 + 16 * q, 256, k0, lane);
      acc[q] = __builtin_amdgcn_wmma_f32_16x16x32_bf16(false, a, false, b,
                                                       (short)0, acc[q], false, false);
    }
  }
  // 512-column boundary is strip-aligned -> wave-uniform destination select.
  float* __restrict__ dst = (n0 < D_IN) ? xm : gate;
  const int ncol = (n0 < D_IN) ? n0 : (n0 - D_IN);
  const int mrb  = m0 + ((lane >> 4) << 3);
#pragma unroll
  for (int q = 0; q < 4; ++q) {
    const int nl = ncol + 16 * q + (lane & 15);
#pragma unroll
    for (int r = 0; r < 8; ++r)
      dst[(size_t)(mrb + r) * D_IN + nl] = acc[q][r];
  }
}

// ---------------------------------------------------------------------------
// Kernel 2: per-direction causal depthwise conv (k=4) + SiLU.
// ---------------------------------------------------------------------------
__global__ void k_conv_silu(const float* __restrict__ xm, const float* __restrict__ conv_w,
                            const float* __restrict__ conv_b,
                            float* __restrict__ xs32, bf16_t* __restrict__ xs16) {
  const size_t idx = (size_t)blockIdx.x * blockDim.x + threadIdx.x;  // 8*4096*512
  const int d   = (int)(idx & 511);
  const int t   = (int)((idx >> 9) & 4095);
  const int g   = (int)(idx >> 21);
  const int dir = g >> 1, b = g & 1;

  float acc = conv_b[d];
#pragma unroll
  for (int k = 0; k < 4; ++k) {
    const int tau = t - 3 + k;
    if (tau >= 0) {
      int h = tau >> 6, w = tau & 63;
      if (dir & 1) w = 63 - w;
      if (dir & 2) h = 63 - h;
      acc = fmaf(conv_w[d * 4 + k],
                 xm[((size_t)b * L_SEQ + ((h << 6) + w)) * D_IN + d], acc);
    }
  }
  const float s = acc * sigmoidf_(acc);
  xs32[idx] = s;
  xs16[idx] = (bf16_t)s;
}

// ---------------------------------------------------------------------------
// Kernel 3: dbl = xs @ W_x.  M=32768, K=512, N=48 (one 16x48 strip per wave).
// ---------------------------------------------------------------------------
__global__ void k_gemm_dbl(const bf16_t* __restrict__ xs16, const bf16_t* __restrict__ W_xT,
                           float* __restrict__ dbl) {
  const int lane = threadIdx.x & 31;
  const int wave = threadIdx.x >> 5;
  const int mt = blockIdx.x * 8 + wave;       // 2048 M tiles
  const int m0 = mt << 4;

  v8f acc[3] = {};
  for (int k0 = 0; k0 < 512; k0 += 32) {
    const v16bf a = load_frag16(xs16, m0, 512, k0, lane);
#pragma unroll
    for (int q = 0; q < 3; ++q) {
      const v16bf b = load_frag16(W_xT, 16 * q, 512, k0, lane);
      acc[q] = __builtin_amdgcn_wmma_f32_16x16x32_bf16(false, a, false, b,
                                                       (short)0, acc[q], false, false);
    }
  }
  const int mrb = m0 + ((lane >> 4) << 3);
#pragma unroll
  for (int q = 0; q < 3; ++q) {
    const int nl = 16 * q + (lane & 15);
#pragma unroll
    for (int r = 0; r < 8; ++r)
      dbl[(size_t)(mrb + r) * DBLW + nl] = acc[q][r];
  }
}

// ---------------------------------------------------------------------------
// Kernel 4: fused selective scan (one thread per (g, d) channel).
// dbl rows staged into LDS via GLOBAL_LOAD_ASYNC_TO_LDS_B32 (ASYNCcnt path).
// ---------------------------------------------------------------------------
#define SCAN_T   128
#define SCAN_CH  8
__global__ void k_scan(const float* __restrict__ dbl, const float* __restrict__ xs32,
                       const float* __restrict__ gate, const float* __restrict__ W_dt,
                       const float* __restrict__ b_dt, const float* __restrict__ A_log,
                       const float* __restrict__ Dp, float* __restrict__ y_all) {
  __shared__ float sdbl[SCAN_CH][DBLW];
  const int g   = blockIdx.x;                          // 0..7
  const int d   = blockIdx.y * SCAN_T + threadIdx.x;   // channel
  const int dir = g >> 1, b = g & 1;

  // Low 32 bits of a generic shared pointer == wave LDS byte offset (ISA 10.2).
  const unsigned int lds0 = (unsigned int)(uintptr_t)(&sdbl[0][0]);

  float Arow[NSTATE], st[NSTATE], wdt[DTR];
#pragma unroll
  for (int s = 0; s < NSTATE; ++s) {
    Arow[s] = -__expf(A_log[d * NSTATE + s]);
    st[s]   = 0.0f;
  }
#pragma unroll
  for (int r = 0; r < DTR; ++r) wdt[r] = W_dt[r * D_IN + d];
  const float bdt = b_dt[d];
  const float Dd  = Dp[d];
  const size_t rowbase = (size_t)g * L_SEQ;

  for (int t0 = 0; t0 < L_SEQ; t0 += SCAN_CH) {
    __syncthreads();                                   // protect sdbl reuse
    // Async cooperative stage of SCAN_CH*48 floats (3 dwords per thread).
    for (int i = threadIdx.x; i < SCAN_CH * DBLW; i += SCAN_T) {
      const unsigned long long ga =
          (unsigned long long)(uintptr_t)(dbl + (rowbase + t0) * DBLW + i);
      const unsigned int la = lds0 + (unsigned int)(i * 4);
      asm volatile("global_load_async_to_lds_b32 %0, %1, off"
                   :: "v"(la), "v"(ga) : "memory");
    }
    if (t0 + SCAN_CH < L_SEQ)
      __builtin_prefetch(&dbl[(rowbase + t0 + SCAN_CH) * DBLW], 0, 0);
    asm volatile("s_wait_asynccnt 0x0" ::: "memory");
    __syncthreads();

#pragma unroll
    for (int tt = 0; tt < SCAN_CH; ++tt) {
      const int t = t0 + tt;
      // dt = softplus(dbl[t,0:16] @ W_dt[:,d] + b_dt[d])
      float xr = bdt;
#pragma unroll
      for (int r = 0; r < DTR; ++r) xr = fmaf(sdbl[tt][r], wdt[r], xr);
      const float dt  = fmaxf(xr, 0.0f) + log1pf(__expf(-fabsf(xr)));
      const float x_t = xs32[(rowbase + t) * D_IN + d];
      const float dx  = dt * x_t;
      float y = 0.0f;
#pragma unroll
      for (int s = 0; s < NSTATE; ++s) {
        const float dA = __expf(dt * Arow[s]);
        st[s] = fmaf(dA, st[s], dx * sdbl[tt][DTR + s]);
        y     = fmaf(st[s], sdbl[tt][DTR + NSTATE + s], y);
      }
      y = fmaf(x_t, Dd, y);
      // gate lives at the un-flipped pixel l = map_dir(t)
      int h = t >> 6, w = t & 63;
      if (dir & 1) w = 63 - w;
      if (dir & 2) h = 63 - h;
      const float gv = gate[((size_t)b * L_SEQ + ((h << 6) + w)) * D_IN + d];
      y *= gv * sigmoidf_(gv);
      y_all[(rowbase + t) * D_IN + d] = y;
    }
  }
}

// ---------------------------------------------------------------------------
// Kernel 5: un-flip + average the 4 directions, emit bf16 for the out-GEMM.
// ---------------------------------------------------------------------------
__global__ void k_combine(const float* __restrict__ y_all, bf16_t* __restrict__ ysum) {
  const size_t idx = (size_t)blockIdx.x * blockDim.x + threadIdx.x;  // 2*4096*512
  const int d = (int)(idx & 511);
  const int l = (int)((idx >> 9) & 4095);
  const int b = (int)(idx >> 21);
  const int h = l >> 6, w = l & 63;
  float acc = 0.0f;
#pragma unroll
  for (int dir = 0; dir < NDIR; ++dir) {
    const int ww = (dir & 1) ? 63 - w : w;
    const int hh = (dir & 2) ? 63 - h : h;
    acc += y_all[(((size_t)(dir * NB + b)) * L_SEQ + ((hh << 6) + ww)) * D_IN + d];
  }
  ysum[idx] = (bf16_t)(acc * 0.25f);
}

// ---------------------------------------------------------------------------
// Kernel 6: out = 0.25*Σ flip(y_dir) @ W_out.  M=8192, K=512, N=256.
// ---------------------------------------------------------------------------
__global__ void k_gemm_out(const bf16_t* __restrict__ ysum, const bf16_t* __restrict__ W_outT,
                           float* __restrict__ out) {
  const int lane = threadIdx.x & 31;
  const int wave = threadIdx.x >> 5;
  const int task = blockIdx.x * 8 + wave;     // 512 Mtiles * 4 strips = 2048
  const int mt = task >> 2;
  const int ns = task & 3;
  const int m0 = mt << 4, n0 = ns << 6;

  v8f acc[4] = {};
  for (int k0 = 0; k0 < 512; k0 += 32) {
    const v16bf a = load_frag16(ysum, m0, 512, k0, lane);
#pragma unroll
    for (int q = 0; q < 4; ++q) {
      const v16bf b = load_frag16(W_outT, n0 + 16 * q, 512, k0, lane);
      acc[q] = __builtin_amdgcn_wmma_f32_16x16x32_bf16(false, a, false, b,
                                                       (short)0, acc[q], false, false);
    }
  }
  const int mrb = m0 + ((lane >> 4) << 3);
#pragma unroll
  for (int q = 0; q < 4; ++q) {
    const int cn = n0 + 16 * q + (lane & 15);
#pragma unroll
    for (int r = 0; r < 8; ++r) {
      const int m = mrb + r;
      out[((size_t)(m >> 12) * 256 + cn) * 4096 + (m & 4095)] = acc[q][r];
    }
  }
}

// ---------------------------------------------------------------------------
extern "C" void kernel_launch(void* const* d_in, const int* in_sizes, int n_in,
                              void* d_out, int out_size, void* d_ws, size_t ws_size,
                              hipStream_t stream) {
  const float* x      = (const float*)d_in[0];
  const float* W_in   = (const float*)d_in[1];
  const float* conv_w = (const float*)d_in[2];
  const float* conv_b = (const float*)d_in[3];
  const float* W_x    = (const float*)d_in[4];
  const float* W_dt   = (const float*)d_in[5];
  const float* b_dt   = (const float*)d_in[6];
  const float* A_log  = (const float*)d_in[7];
  const float* Dp     = (const float*)d_in[8];
  const float* W_out  = (const float*)d_in[9];
  float* out = (float*)d_out;

  char* ws = (char*)d_ws;
  float*  xm     = (float*)(ws + OFF_XM);
  float*  gate   = (float*)(ws + OFF_GATE);
  float*  xs32   = (float*)(ws + OFF_XS32);
  bf16_t* xs16   = (bf16_t*)(ws + OFF_XS16);
  float*  dbl    = (float*)(ws + OFF_DBL);
  float*  y_all  = (float*)(ws + OFF_YALL);
  bf16_t* ysum   = (bf16_t*)(ws + OFF_YSUM);
  bf16_t* xT     = (bf16_t*)(ws + OFF_XT);
  bf16_t* W_inT  = (bf16_t*)(ws + OFF_WINT);
  bf16_t* W_xT   = (bf16_t*)(ws + OFF_WXT);
  bf16_t* W_outT = (bf16_t*)(ws + OFF_WOUTT);

  // 0) pack operands to bf16 in fragment-friendly layouts
  k_prep_xT<<<8192, 256, 0, stream>>>(x, xT);
  k_prep_wT<<<1024, 256, 0, stream>>>(W_in, W_inT, 256, 1024);
  k_prep_wT<<<96,   256, 0, stream>>>(W_x, W_xT, 512, DBLW);
  k_prep_wT<<<512,  256, 0, stream>>>(W_out, W_outT, 512, 256);
  // 1) input projection (once for all 4 directions)
  k_gemm_in<<<1024, 256, 0, stream>>>(xT, W_inT, xm, gate);
  // 2) per-direction depthwise conv + SiLU
  k_conv_silu<<<65536, 256, 0, stream>>>(xm, conv_w, conv_b, xs32, xs16);
  // 3) dbl projection (dt_raw | B | C)
  k_gemm_dbl<<<256, 256, 0, stream>>>(xs16, W_xT, dbl);
  // 4) fused selective scan (dt-proj + recurrence + epilogue)
  k_scan<<<dim3(8, 4), SCAN_T, 0, stream>>>(dbl, xs32, gate, W_dt, b_dt, A_log, Dp, y_all);
  // 5) un-flip + average
  k_combine<<<16384, 256, 0, stream>>>(y_all, ysum);
  // 6) output projection (once)
  k_gemm_out<<<256, 256, 0, stream>>>(ysum, W_outT, out);
}
